// NConv_89163521065157
// MI455X (gfx1250) — matrix operand order repaired and checked by reading, not compiled, not run on hardware
//
#include <hip/hip_runtime.h>

#define N_NODES 50000
#define DIM     64
#define N_EDGES 800000
#define NB      50
#define NMAXR   1000
#define NCL     100

typedef __attribute__((ext_vector_type(16))) __bf16 v16bf;
typedef __attribute__((ext_vector_type(8)))  __bf16 v8bf;
typedef __attribute__((ext_vector_type(8)))  float  v8f;
typedef __attribute__((ext_vector_type(4)))  float  v4f;

union ABFrag  { v16bf v; v8bf h[2]; };
union ABBytes { v16bf v; __bf16 b[16]; };

__device__ __forceinline__ v8f wmma_bf16(v16bf a, v16bf b, v8f c) {
  return __builtin_amdgcn_wmma_f32_16x16x32_bf16(false, a, false, b, (short)0, c,
                                                 false, false);
}

// ---- A fragment (16x32 bf16, lane = row) built DIRECTLY from a global fp32 row.
// lanes 0-15:  elems 0-7 -> K kbase+0..7,   elems 8-15 -> K kbase+16..23
// lanes 16-31: elems 0-7 -> K kbase+8..15,  elems 8-15 -> K kbase+24..31
__device__ __forceinline__ v16bf make_A_f32(const float* __restrict__ row,
                                            int kbase, int hi) {
  const float* p0 = row + kbase + hi * 8;
  const float* p1 = row + kbase + 16 + hi * 8;
  v4f f0 = *(const v4f*)p0, f1 = *(const v4f*)(p0 + 4);
  v4f f2 = *(const v4f*)p1, f3 = *(const v4f*)(p1 + 4);
  ABBytes u;
  u.b[0]  = (__bf16)f0.x; u.b[1]  = (__bf16)f0.y; u.b[2]  = (__bf16)f0.z; u.b[3]  = (__bf16)f0.w;
  u.b[4]  = (__bf16)f1.x; u.b[5]  = (__bf16)f1.y; u.b[6]  = (__bf16)f1.z; u.b[7]  = (__bf16)f1.w;
  u.b[8]  = (__bf16)f2.x; u.b[9]  = (__bf16)f2.y; u.b[10] = (__bf16)f2.z; u.b[11] = (__bf16)f2.w;
  u.b[12] = (__bf16)f3.x; u.b[13] = (__bf16)f3.y; u.b[14] = (__bf16)f3.z; u.b[15] = (__bf16)f3.w;
  return u.v;
}

// A fragment of bf16(a[k] + ep*x[k]) built from two global fp32 rows.
__device__ __forceinline__ v16bf make_A_combine(const float* __restrict__ a,
                                                const float* __restrict__ xr,
                                                float ep, int kbase, int hi) {
  int o0 = kbase + hi * 8, o1 = kbase + 16 + hi * 8;
  v4f a0 = *(const v4f*)(a + o0), a1 = *(const v4f*)(a + o0 + 4);
  v4f a2 = *(const v4f*)(a + o1), a3 = *(const v4f*)(a + o1 + 4);
  v4f x0 = *(const v4f*)(xr + o0), x1 = *(const v4f*)(xr + o0 + 4);
  v4f x2 = *(const v4f*)(xr + o1), x3 = *(const v4f*)(xr + o1 + 4);
  ABBytes u;
  u.b[0]  = (__bf16)(a0.x + ep * x0.x); u.b[1]  = (__bf16)(a0.y + ep * x0.y);
  u.b[2]  = (__bf16)(a0.z + ep * x0.z); u.b[3]  = (__bf16)(a0.w + ep * x0.w);
  u.b[4]  = (__bf16)(a1.x + ep * x1.x); u.b[5]  = (__bf16)(a1.y + ep * x1.y);
  u.b[6]  = (__bf16)(a1.z + ep * x1.z); u.b[7]  = (__bf16)(a1.w + ep * x1.w);
  u.b[8]  = (__bf16)(a2.x + ep * x2.x); u.b[9]  = (__bf16)(a2.y + ep * x2.y);
  u.b[10] = (__bf16)(a2.z + ep * x2.z); u.b[11] = (__bf16)(a2.w + ep * x2.w);
  u.b[12] = (__bf16)(a3.x + ep * x3.x); u.b[13] = (__bf16)(a3.y + ep * x3.y);
  u.b[14] = (__bf16)(a3.z + ep * x3.z); u.b[15] = (__bf16)(a3.w + ep * x3.w);
  return u.v;
}

// Tail A fragment for k_pool: kbase=96, only k<100 valid (K padded 100->128).
__device__ __forceinline__ v16bf make_A_tail(const float* __restrict__ row, int hi) {
  ABBytes u;
#pragma unroll
  for (int i = 0; i < 8; ++i) {
    int k0 = 96 + hi * 8 + i;
    u.b[i] = (k0 < NCL) ? (__bf16)row[k0] : (__bf16)0.f;
    u.b[8 + i] = (__bf16)0.f;  // k >= 112 always padding
  }
  return u.v;
}

__device__ __forceinline__ v16bf zero_frag() {
  ABBytes u;
#pragma unroll
  for (int i = 0; i < 16; ++i) u.b[i] = (__bf16)0.f;
  return u.v;
}

// A fragment from bf16 tile in LDS (used only for the h1|h2 transpose in k_out).
__device__ __forceinline__ v16bf load_A_frag(const __bf16* base, int stride,
                                             int kbase, int lane) {
  int row = lane & 15, hi = lane >> 4;
  const __bf16* p = base + row * stride + kbase + hi * 8;
  ABFrag u;
  u.h[0] = *(const v8bf*)p;
  u.h[1] = *(const v8bf*)(p + 16);
  return u.v;
}

// B fragment (32x16 KxN) from N-major (transposed) bf16 storage wT[n*strideK+k].
__device__ __forceinline__ v16bf load_B_frag(const __bf16* wT, int strideK,
                                             int nbase, int kbase, int lane) {
  int n = nbase + (lane & 15), hi = lane >> 4;
  const __bf16* p = wT + (size_t)n * strideK + kbase + hi * 8;
  ABFrag u;
  u.h[0] = *(const v8bf*)p;
  u.h[1] = *(const v8bf*)(p + 16);
  return u.v;
}

__global__ void __launch_bounds__(256) k_zero(float* p, int n) {
  int i = blockIdx.x * blockDim.x + threadIdx.x;
  int s = gridDim.x * blockDim.x;
  for (; i < n; i += s) p[i] = 0.f;
}

// ---- Kernel 1: msg_up = relu(attr1 @ W_mu + b_mu) stored transposed+padded:
//      msg_upT[b][n(64)][cpad(128)] bf16; c>=100 stays zero (pre-zeroed).
__global__ void __launch_bounds__(256) k_cmlp(const float* __restrict__ attr1,
                                              const float* __restrict__ W_mu,
                                              const float* __restrict__ b_mu,
                                              __bf16* __restrict__ msg_upT) {
  __shared__ __bf16 sW[64 * 64];  // W_mu transposed: [n][k]
  int tid = threadIdx.x, wave = tid >> 5, lane = tid & 31;
  int hi = lane >> 4, lo = lane & 15;
  for (int i = tid; i < 64 * 64; i += 256) {
    int n = i >> 6, k = i & 63;
    sW[i] = (__bf16)W_mu[k * 64 + n];
  }
  __syncthreads();
  const int NT = (NB * NCL + 15) / 16;  // 313
  for (int t = blockIdx.x * 8 + wave; t < NT; t += gridDim.x * 8) {
    int row = t * 16 + lo;
    v16bf a0, a1;
    if (row < NB * NCL) {
      const float* src = attr1 + (size_t)row * 64;
      a0 = make_A_f32(src, 0, hi);
      a1 = make_A_f32(src, 32, hi);
    } else {
      a0 = zero_frag();
      a1 = zero_frag();
    }
#pragma unroll
    for (int nt = 0; nt < 4; ++nt) {
      float bias = b_mu[nt * 16 + lo];
      v8f acc;
#pragma unroll
      for (int r = 0; r < 8; ++r) acc[r] = bias;
      acc = wmma_bf16(a0, load_B_frag(sW, 64, nt * 16, 0, lane), acc);
      acc = wmma_bf16(a1, load_B_frag(sW, 64, nt * 16, 32, lane), acc);
#pragma unroll
      for (int r = 0; r < 8; ++r) {
        int rr = t * 16 + r + hi * 8;
        if (rr < NB * NCL) {
          int b = rr / NCL, c = rr - b * NCL;
          float v = acc[r] > 0.f ? acc[r] : 0.f;
          msg_upT[((size_t)(b * 64 + nt * 16 + lo)) * 128 + c] = (__bf16)v;
        }
      }
    }
  }
}

// ---- Kernel 2 (dominant): edge MLP + atomic scatter-sum into agg_adj.
//      A fragments built directly from gathered x rows and streamed e rows.
__global__ void __launch_bounds__(256) k_edge(const float* __restrict__ x,
                                              const float* __restrict__ e,
                                              const int* __restrict__ edge_index,
                                              const float* __restrict__ W_ma,
                                              const float* __restrict__ b_ma,
                                              float* __restrict__ agg_adj) {
  __shared__ __bf16 sW[64 * 128];  // W_ma transposed: [n(64)][k(128)]
  int tid = threadIdx.x, wave = tid >> 5, lane = tid & 31;
  int hi = lane >> 4, lo = lane & 15;
  for (int i = tid; i < 64 * 128; i += 256) {
    int n = i >> 7, k = i & 127;
    sW[i] = (__bf16)W_ma[k * 64 + n];
  }
  __syncthreads();
  const int NT = N_EDGES / 16;  // 50000
  for (int t = blockIdx.x * 8 + wave; t < NT; t += gridDim.x * 8) {
    int eid0 = t * 16;
    int src = edge_index[eid0 + lo];                 // gather source node
    const float* xrow = x + (size_t)src * 64;
    const float* erow = e + (size_t)(eid0 + lo) * 64;
    int dsts[8];
#pragma unroll
    for (int r = 0; r < 8; ++r)
      dsts[r] = edge_index[N_EDGES + eid0 + hi * 8 + r];
    // features 0..63 = x_j, 64..127 = e  (matches concat([x_j, e]) @ W_ma)
    v16bf a0 = make_A_f32(xrow, 0, hi);
    v16bf a1 = make_A_f32(xrow, 32, hi);
    v16bf a2 = make_A_f32(erow, 0, hi);
    v16bf a3 = make_A_f32(erow, 32, hi);
#pragma unroll
    for (int nt = 0; nt < 4; ++nt) {
      float bias = b_ma[nt * 16 + lo];
      v8f acc;
#pragma unroll
      for (int r = 0; r < 8; ++r) acc[r] = bias;
      acc = wmma_bf16(a0, load_B_frag(sW, 128, nt * 16, 0, lane), acc);
      acc = wmma_bf16(a1, load_B_frag(sW, 128, nt * 16, 32, lane), acc);
      acc = wmma_bf16(a2, load_B_frag(sW, 128, nt * 16, 64, lane), acc);
      acc = wmma_bf16(a3, load_B_frag(sW, 128, nt * 16, 96, lane), acc);
#pragma unroll
      for (int r = 0; r < 8; ++r) {
        float v = acc[r] > 0.f ? acc[r] : 0.f;  // relu BEFORE segment-sum
        atomicAdd(agg_adj + (size_t)dsts[r] * 64 + nt * 16 + lo, v);
      }
    }
  }
}

// ---- Kernel 3: agg_up[b] = nc1[b] (1000x100, K padded to 128) @ msg_up[b]
__global__ void __launch_bounds__(256) k_pool(const float* __restrict__ nc1,
                                              const __bf16* __restrict__ msg_upT,
                                              float* __restrict__ agg_up) {
  int tid = threadIdx.x, wave = tid >> 5, lane = tid & 31;
  int hi = lane >> 4, lo = lane & 15;
  const int TPB = (NMAXR + 15) / 16;  // 63
  const int NT = NB * TPB;
  for (int t = blockIdx.x * 8 + wave; t < NT; t += gridDim.x * 8) {
    int bi = t / TPB, row0 = (t % TPB) * 16;
    int row = row0 + lo;
    v16bf a0, a1, a2, a3;
    if (row < NMAXR) {
      const float* r = nc1 + ((size_t)(bi * NMAXR + row)) * NCL;
      a0 = make_A_f32(r, 0, hi);
      a1 = make_A_f32(r, 32, hi);
      a2 = make_A_f32(r, 64, hi);   // k 64..95 all < 100
      a3 = make_A_tail(r, hi);      // k 96..127, only k<100 valid
    } else {
      a0 = zero_frag(); a1 = zero_frag(); a2 = zero_frag(); a3 = zero_frag();
    }
    const __bf16* wB = msg_upT + (size_t)bi * 64 * 128;
#pragma unroll
    for (int nt = 0; nt < 4; ++nt) {
      v8f acc;
#pragma unroll
      for (int r = 0; r < 8; ++r) acc[r] = 0.f;
      acc = wmma_bf16(a0, load_B_frag(wB, 128, nt * 16, 0, lane), acc);
      acc = wmma_bf16(a1, load_B_frag(wB, 128, nt * 16, 32, lane), acc);
      acc = wmma_bf16(a2, load_B_frag(wB, 128, nt * 16, 64, lane), acc);
      acc = wmma_bf16(a3, load_B_frag(wB, 128, nt * 16, 96, lane), acc);
#pragma unroll
      for (int r = 0; r < 8; ++r) {
        int rr = row0 + r + hi * 8;
        if (rr < NMAXR)
          agg_up[((size_t)(bi * NMAXR + rr)) * 64 + nt * 16 + lo] = acc[r];
      }
    }
  }
}

// ---- Kernel 4: fused node update + output projection.
//      LDS used only for the D-layout -> A-layout transpose of h1|h2.
__global__ void __launch_bounds__(256) k_out(const float* __restrict__ x,
                                             const float* __restrict__ agg_adj,
                                             const float* __restrict__ agg_up,
                                             const int* __restrict__ x_idx,
                                             const float* __restrict__ eps1,
                                             const float* __restrict__ eps2,
                                             const float* __restrict__ W_ua,
                                             const float* __restrict__ b_ua,
                                             const float* __restrict__ W_uu,
                                             const float* __restrict__ b_uu,
                                             const float* __restrict__ W_c,
                                             const float* __restrict__ b_c,
                                             float* __restrict__ out) {
  __shared__ __bf16 sWua[64 * 64];
  __shared__ __bf16 sWuu[64 * 64];
  __shared__ __bf16 sWc[64 * 128];
  __shared__ __bf16 sH[8][16 * 128];
  int tid = threadIdx.x, wave = tid >> 5, lane = tid & 31;
  int hi = lane >> 4, lo = lane & 15;
  for (int i = tid; i < 64 * 64; i += 256) {
    int n = i >> 6, k = i & 63;
    sWua[i] = (__bf16)W_ua[k * 64 + n];
    sWuu[i] = (__bf16)W_uu[k * 64 + n];
  }
  for (int i = tid; i < 64 * 128; i += 256) {
    int n = i >> 7, k = i & 127;
    sWc[i] = (__bf16)W_c[k * 64 + n];
  }
  __syncthreads();
  float ep1 = 1.f + eps1[0], ep2 = 1.f + eps2[0];
  const int NT = N_NODES / 16;  // 3125
  for (int t = blockIdx.x * 8 + wave; t < NT; t += gridDim.x * 8) {
    int node = t * 16 + lo;
    const float* pa1 = agg_adj + (size_t)node * 64;
    const float* px  = x + (size_t)node * 64;
    v16bf u0 = make_A_combine(pa1, px, ep1, 0, hi);
    v16bf u1 = make_A_combine(pa1, px, ep1, 32, hi);
#pragma unroll
    for (int nt = 0; nt < 4; ++nt) {  // h1 = relu(.. @ W_ua + b_ua) -> sH cols 0..63
      float bias = b_ua[nt * 16 + lo];
      v8f acc;
#pragma unroll
      for (int r = 0; r < 8; ++r) acc[r] = bias;
      acc = wmma_bf16(u0, load_B_frag(sWua, 64, nt * 16, 0, lane), acc);
      acc = wmma_bf16(u1, load_B_frag(sWua, 64, nt * 16, 32, lane), acc);
#pragma unroll
      for (int r = 0; r < 8; ++r)
        sH[wave][(r + hi * 8) * 128 + nt * 16 + lo] =
            (__bf16)(acc[r] > 0.f ? acc[r] : 0.f);
    }
    int nodeUp = x_idx[node];
    const float* pa2 = agg_up + (size_t)nodeUp * 64;
    v16bf w0 = make_A_combine(pa2, px, ep2, 0, hi);
    v16bf w1 = make_A_combine(pa2, px, ep2, 32, hi);
#pragma unroll
    for (int nt = 0; nt < 4; ++nt) {  // h2 -> sH cols 64..127
      float bias = b_uu[nt * 16 + lo];
      v8f acc;
#pragma unroll
      for (int r = 0; r < 8; ++r) acc[r] = bias;
      acc = wmma_bf16(w0, load_B_frag(sWuu, 64, nt * 16, 0, lane), acc);
      acc = wmma_bf16(w1, load_B_frag(sWuu, 64, nt * 16, 32, lane), acc);
#pragma unroll
      for (int r = 0; r < 8; ++r)
        sH[wave][(r + hi * 8) * 128 + 64 + nt * 16 + lo] =
            (__bf16)(acc[r] > 0.f ? acc[r] : 0.f);
    }
    // out = [h1|h2] @ W_c + b_c   (A fragments re-read from LDS: layout transpose)
    v16bf h0  = load_A_frag(sH[wave], 128, 0, lane);
    v16bf h1f = load_A_frag(sH[wave], 128, 32, lane);
    v16bf h2f = load_A_frag(sH[wave], 128, 64, lane);
    v16bf h3f = load_A_frag(sH[wave], 128, 96, lane);
#pragma unroll
    for (int nt = 0; nt < 4; ++nt) {
      float bias = b_c[nt * 16 + lo];
      v8f acc;
#pragma unroll
      for (int r = 0; r < 8; ++r) acc[r] = bias;
      acc = wmma_bf16(h0,  load_B_frag(sWc, 128, nt * 16, 0, lane), acc);
      acc = wmma_bf16(h1f, load_B_frag(sWc, 128, nt * 16, 32, lane), acc);
      acc = wmma_bf16(h2f, load_B_frag(sWc, 128, nt * 16, 64, lane), acc);
      acc = wmma_bf16(h3f, load_B_frag(sWc, 128, nt * 16, 96, lane), acc);
#pragma unroll
      for (int r = 0; r < 8; ++r)
        out[(size_t)(t * 16 + r + hi * 8) * 64 + nt * 16 + lo] = acc[r];
    }
  }
}

extern "C" void kernel_launch(void* const* d_in, const int* in_sizes, int n_in,
                              void* d_out, int out_size, void* d_ws,
                              size_t ws_size, hipStream_t stream) {
  (void)in_sizes; (void)n_in; (void)out_size; (void)ws_size;
  const float* x        = (const float*)d_in[0];
  const float* e        = (const float*)d_in[1];
  const int*   edge_idx = (const int*)d_in[2];
  const float* attr1    = (const float*)d_in[3];
  const float* nc1      = (const float*)d_in[4];
  const int*   x_idx    = (const int*)d_in[5];
  const float* eps1     = (const float*)d_in[6];
  const float* eps2     = (const float*)d_in[7];
  const float* W_ma = (const float*)d_in[8];  const float* b_ma = (const float*)d_in[9];
  const float* W_mu = (const float*)d_in[10]; const float* b_mu = (const float*)d_in[11];
  const float* W_ua = (const float*)d_in[12]; const float* b_ua = (const float*)d_in[13];
  const float* W_uu = (const float*)d_in[14]; const float* b_uu = (const float*)d_in[15];
  const float* W_c  = (const float*)d_in[16]; const float* b_c  = (const float*)d_in[17];

  char* ws = (char*)d_ws;
  float*  agg_adj = (float*)ws;                 // N*64*4  = 12,800,000 B
  float*  agg_up  = (float*)(ws + 12800000);    // N*64*4  = 12,800,000 B
  __bf16* msg_upT = (__bf16*)(ws + 25600000);   // 50*64*128*2 = 819,200 B

  k_zero<<<400, 256, 0, stream>>>(agg_adj, N_NODES * 64);
  k_zero<<<100, 256, 0, stream>>>((float*)msg_upT, NB * 64 * 128 / 2);
  k_cmlp<<<40, 256, 0, stream>>>(attr1, W_mu, b_mu, msg_upT);
  k_edge<<<1024, 256, 0, stream>>>(x, e, edge_idx, W_ma, b_ma, agg_adj);
  k_pool<<<200, 256, 0, stream>>>(nc1, msg_upT, agg_up);
  k_out<<<200, 256, 0, stream>>>(x, agg_adj, agg_up, x_idx, eps1, eps2, W_ua,
                                 b_ua, W_uu, b_uu, W_c, b_c, (float*)d_out);
}
